// SynthesizerEncoder_13280038879459
// MI455X (gfx1250) — matrix operand order repaired and checked by reading, not compiled, not run on hardware
//
#include <hip/hip_runtime.h>
#include <cmath>

typedef __bf16 bf16_t;
typedef __attribute__((ext_vector_type(16))) __bf16 v16bf;
typedef __attribute__((ext_vector_type(8)))  float  v8f;

#define DEV __device__ __forceinline__

DEV v8f wmma_bf16(v16bf a, v16bf b, v8f c) {
    return __builtin_amdgcn_wmma_f32_16x16x32_bf16(false, a, false, b, (short)0, c, false, false);
}

// ---- CDNA5 async global->LDS copy (ASYNCcnt-tracked DMA, no VGPR data path) ----
// LDS byte offset = low 32 bits of the generic pointer (ISA: LDS flat addr = {aperture, off[31:0]})
DEV unsigned lds_off_of(const void* p) { return (unsigned)(unsigned long long)p; }
DEV void async_copy_b128(unsigned lds_off, const void* gptr) {
    asm volatile("global_load_async_to_lds_b128 %0, %1, off"
                 :: "v"(lds_off), "v"(gptr)
                 : "memory");
}
DEV void wait_async0() { asm volatile("s_wait_asynccnt 0x0" ::: "memory"); }

// ---- 16-lane butterfly reductions: DPP fused directly into max/add (1 instr/step) ----
DEV float dpp_max16(float x) {
    asm("v_max_num_f32_dpp %0, %0, %0 quad_perm:[1,0,3,2] row_mask:0xf bank_mask:0xf bound_ctrl:1" : "+v"(x));
    asm("v_max_num_f32_dpp %0, %0, %0 quad_perm:[2,3,0,1] row_mask:0xf bank_mask:0xf bound_ctrl:1" : "+v"(x));
    asm("v_max_num_f32_dpp %0, %0, %0 row_half_mirror row_mask:0xf bank_mask:0xf bound_ctrl:1" : "+v"(x));
    asm("v_max_num_f32_dpp %0, %0, %0 row_mirror row_mask:0xf bank_mask:0xf bound_ctrl:1" : "+v"(x));
    return x;
}
DEV float dpp_sum16(float x) {
    asm("v_add_f32_dpp %0, %0, %0 quad_perm:[1,0,3,2] row_mask:0xf bank_mask:0xf bound_ctrl:1" : "+v"(x));
    asm("v_add_f32_dpp %0, %0, %0 quad_perm:[2,3,0,1] row_mask:0xf bank_mask:0xf bound_ctrl:1" : "+v"(x));
    asm("v_add_f32_dpp %0, %0, %0 row_half_mirror row_mask:0xf bank_mask:0xf bound_ctrl:1" : "+v"(x));
    asm("v_add_f32_dpp %0, %0, %0 row_mirror row_mask:0xf bank_mask:0xf bound_ctrl:1" : "+v"(x));
    return x;
}

// ---- fp32 -> bf16 (vectorized, n % 4 == 0) ----
__global__ void cvt_bf16_kernel(const float* __restrict__ s, bf16_t* __restrict__ d, int n) {
    const int i = (blockIdx.x * blockDim.x + threadIdx.x) * 4;
    if (i < n) {
        const float4 v = *(const float4*)(s + i);
        bf16_t o[4] = {(bf16_t)v.x, (bf16_t)v.y, (bf16_t)v.z, (bf16_t)v.w};
        *(uint2*)(d + i) = *(const uint2*)o;
    }
}

// ---- fp32 [K][N] -> bf16 [N][K] (K-major for B operands); K,N % 32 == 0 ----
__global__ __launch_bounds__(256)
void cvtT_bf16_kernel(const float* __restrict__ src, bf16_t* __restrict__ dst, int K, int N) {
    __shared__ float tile[32][33];
    const int tx = threadIdx.x & 31, ty = threadIdx.x >> 5;
    const int k0 = blockIdx.y * 32, n0 = blockIdx.x * 32;
#pragma unroll
    for (int i = 0; i < 4; ++i)
        tile[ty + 8 * i][tx] = src[(long long)(k0 + ty + 8 * i) * N + n0 + tx];
    __syncthreads();
#pragma unroll
    for (int i = 0; i < 4; ++i)
        dst[(long long)(n0 + ty + 8 * i) * K + k0 + tx] = (bf16_t)tile[tx][ty + 8 * i];
}

// ---- bf16 WMMA GEMM, double-buffered LDS fed by async global->LDS DMA ----
// A row-major [M][K]; B K-MAJOR (Bt: [N][K]). 256 thr = 8 waves.
// Block tile 256(M) x 64(N), K step 32; wave owns 32 rows x 64 cols (8 WMMA/step).
// LDS in fragment order: every fragment is one contiguous 32B ds read.
template<int ACT, bool HAS_BIAS, bool HAS_RES, bool OUT_F32, bool OUT_BF16, bool OUT_TRANS>
__global__ __launch_bounds__(256)
void gemm_bf16_kernel(const bf16_t* __restrict__ A, long long a_batch, int lda,
                      const bf16_t* __restrict__ Bt, long long b_batch, int ldk,
                      const float* __restrict__ bias,
                      const float* __restrict__ res, long long res_batch, int ldres,
                      float* __restrict__ Cf, bf16_t* __restrict__ Cb,
                      long long c_batch, int ldc, int K) {
    __shared__ __align__(32) bf16_t As[2][16 * 32 * 16];
    __shared__ __align__(32) bf16_t Bs[2][4 * 32 * 16];
    const int tid  = threadIdx.x;
    const int wave = tid >> 5, lane = tid & 31;
    const int mblk = blockIdx.y * 256;
    const int n0   = blockIdx.x * 64;
    const long long z = blockIdx.z;
    A  += z * a_batch;
    Bt += z * b_batch;

    // hoisted staging offsets
    long long aoff[4]; unsigned as_off[2][4];
#pragma unroll
    for (int u = 0; u < 4; ++u) {
        const int c = tid + u * 256;
        const int g = c >> 6, r = c & 63, sl = r >> 1, jh = (r & 1) << 3;
        const int hi = sl >> 4, M = sl & 15;
        aoff[u] = (long long)(mblk + g * 16 + M) * lda + (jh ? 16 : 0) + 8 * hi;
        const int asl = (g * 32 + sl) * 16 + jh;
        as_off[0][u] = lds_off_of(&As[0][asl]);
        as_off[1][u] = lds_off_of(&As[1][asl]);
    }
    long long boff; unsigned bs_off[2];
    {
        const int t = tid >> 6, r = tid & 63, sl = r >> 1, ih = (r & 1) << 3;
        const int hi = sl >> 4, nl = sl & 15;
        boff = (long long)(n0 + t * 16 + nl) * ldk + (ih ? 16 : 0) + 8 * hi;
        const int bsl = (t * 32 + sl) * 16 + ih;
        bs_off[0] = lds_off_of(&Bs[0][bsl]);
        bs_off[1] = lds_off_of(&Bs[1][bsl]);
    }
    const int a0sl = ((2 * wave) * 32 + lane) * 16;
    const int a1sl = ((2 * wave + 1) * 32 + lane) * 16;

    // kick off tile 0 via async DMA
#pragma unroll
    for (int u = 0; u < 4; ++u) async_copy_b128(as_off[0][u], A + aoff[u]);
    async_copy_b128(bs_off[0], Bt + boff);
    wait_async0();
    __syncthreads();

    v8f acc[2][4] = {};
    const int nk = K >> 5;
    for (int i = 0; i < nk; ++i) {
        const int cur = i & 1;
        const bool more = (i + 1) < nk;
        if (more) {  // async-stage next tile into the other buffer (overlaps with WMMAs)
            const long long k1 = (long long)(i + 1) << 5;
#pragma unroll
            for (int u = 0; u < 4; ++u) async_copy_b128(as_off[cur ^ 1][u], A + aoff[u] + k1);
            async_copy_b128(bs_off[cur ^ 1], Bt + boff + k1);
        }
        const v16bf a0 = *(const v16bf*)(&As[cur][a0sl]);
        const v16bf a1 = *(const v16bf*)(&As[cur][a1sl]);
#pragma unroll
        for (int t = 0; t < 4; ++t) {
            const v16bf b = *(const v16bf*)(&Bs[cur][(t * 32 + lane) * 16]);
            acc[0][t] = wmma_bf16(a0, b, acc[0][t]);
            acc[1][t] = wmma_bf16(a1, b, acc[1][t]);
        }
        if (more) wait_async0();
        __syncthreads();
    }
    // epilogue: VGPR r: M = r + 8*(lane>=16), N = lane&15
    const int mhalf = (lane & 16) ? 8 : 0;
    const int ncol  = lane & 15;
    float*  cf = Cf ? Cf + z * c_batch : nullptr;
    bf16_t* cb = Cb ? Cb + z * c_batch : nullptr;
    const float* rr = HAS_RES ? res + z * res_batch : nullptr;
#pragma unroll
    for (int u = 0; u < 2; ++u)
#pragma unroll
        for (int t = 0; t < 4; ++t) {
            const int n = n0 + t * 16 + ncol;
            const float bv = HAS_BIAS ? bias[n] : 0.f;
#pragma unroll
            for (int r = 0; r < 8; ++r) {
                const long long m = mblk + (2 * wave + u) * 16 + r + mhalf;
                float v = acc[u][t][r] + bv;
                if (ACT == 1) v = v > 0.f ? v : 0.f;
                if (ACT == 2) v = 0.5f * v * (1.f + erff(v * 0.70710678118654752f));
                if (HAS_RES)  v += rr[m * ldres + n];
                if (OUT_F32)  cf[m * ldc + n] = v;
                if (OUT_BF16) {
                    if (OUT_TRANS) cb[(long long)n * ldc + m] = (bf16_t)v;
                    else           cb[m * ldc + n] = (bf16_t)v;
                }
            }
        }
}

// ---- Fused synthesizer attention (flash-style online softmax, async double-buffered) ----
// hidden: [H][4096][64] row-major; w2t: [2048][64] K-major; vht: [H][64][4096] K-major.
__global__ __launch_bounds__(256)
void synth_attn_kernel(const bf16_t* __restrict__ hidden,
                       const bf16_t* __restrict__ w2t,
                       const float*  __restrict__ b2,
                       const bf16_t* __restrict__ vht,
                       bf16_t* __restrict__ ctx_out) {
    __shared__ __align__(32) bf16_t Hs[2 * 8 * 32 * 16];      // A-frag order (16KB)
    __shared__ __align__(32) bf16_t W2s[2][2 * 4 * 32 * 16];  // B-frag order, 2 buf
    __shared__ __align__(32) bf16_t Vs[2][2 * 4 * 32 * 16];   // B-frag order, 2 buf
    __shared__ __align__(32) bf16_t Ps[8][2 * 32 * 16];       // per-wave P, A-frag order
    const int tid = threadIdx.x, wave = tid >> 5, lane = tid & 31;
    const int h    = blockIdx.y;
    const int row0 = blockIdx.x * 128;
    const long long hrow = (long long)h * 4096 * 64;
    const long long hcol = (long long)h * 64 * 4096;
    const int mbase = (row0 >= 2048) ? 2048 : 0;
    const int mhalf = (lane & 16) ? 8 : 0;
    const int ncol  = lane & 15;

    // async-stage hidden tile [128][64] in A-frag order (once)
#pragma unroll
    for (int u = 0; u < 4; ++u) {
        const int c = tid + u * 256;
        const int kc = c >> 9, rem = c & 511;
        const int g = rem >> 6, r = rem & 63, sl = r >> 1, jh = (r & 1) << 3;
        const int hi = sl >> 4, M = sl & 15;
        async_copy_b128(
            lds_off_of(&Hs[((kc * 8 + g) * 32 + sl) * 16 + jh]),
            &hidden[hrow + (long long)(row0 + g * 16 + M) * 64 + kc * 32 + (jh ? 16 : 0) + 8 * hi]);
    }
    // hoisted W2/V staging offsets
    long long w2off[2], voff[2]; unsigned ws_off[2][2], vs_off[2][2];
#pragma unroll
    for (int u = 0; u < 2; ++u) {
        const int c = tid + u * 256;
        const int kc = c >> 8, rem = c & 255;
        const int t = rem >> 6, r = rem & 63, sl = r >> 1, ih = (r & 1) << 3;
        const int hi = sl >> 4, nl = sl & 15;
        const int koff = kc * 32 + (ih ? 16 : 0) + 8 * hi;
        w2off[u] = (long long)(t * 16 + nl) * 64 + koff;
        voff[u]  = hcol + (long long)(t * 16 + nl) * 4096 + mbase + koff;
        const int wsl = ((kc * 4 + t) * 32 + sl) * 16 + ih;
        ws_off[0][u] = lds_off_of(&W2s[0][wsl]);
        ws_off[1][u] = lds_off_of(&W2s[1][wsl]);
        vs_off[0][u] = lds_off_of(&Vs[0][wsl]);
        vs_off[1][u] = lds_off_of(&Vs[1][wsl]);
    }
#pragma unroll
    for (int u = 0; u < 2; ++u) {
        async_copy_b128(ws_off[0][u], w2t + w2off[u]);
        async_copy_b128(vs_off[0][u], vht + voff[u]);
    }
    wait_async0();
    __syncthreads();
    const v16bf aH0 = *(const v16bf*)(&Hs[((0 * 8 + wave) * 32 + lane) * 16]);
    const v16bf aH1 = *(const v16bf*)(&Hs[((1 * 8 + wave) * 32 + lane) * 16]);

    float run_max[8], run_sum[8];
    v8f ctx[4] = {};
#pragma unroll
    for (int r = 0; r < 8; ++r) { run_max[r] = -3.0e38f; run_sum[r] = 0.f; }

    for (int it = 0; it < 32; ++it) {
        const int cur = it & 1;
        const bool more = (it + 1) < 32;
        const int m0 = it * 64;
        if (more) {  // async-stage next slice
            const long long m1 = (long long)(it + 1) * 64;
#pragma unroll
            for (int u = 0; u < 2; ++u) {
                async_copy_b128(ws_off[cur ^ 1][u], w2t + m1 * 64 + w2off[u]);
                async_copy_b128(vs_off[cur ^ 1][u], vht + voff[u] + m1);
            }
        }
        // scores tile [16 x 64] per wave
        v8f s[4];
#pragma unroll
        for (int t = 0; t < 4; ++t) {
            const v16bf b0 = *(const v16bf*)(&W2s[cur][((0 * 4 + t) * 32 + lane) * 16]);
            const v16bf b1 = *(const v16bf*)(&W2s[cur][((1 * 4 + t) * 32 + lane) * 16]);
            v8f sc = {};
            sc = wmma_bf16(aH0, b0, sc);
            sc = wmma_bf16(aH1, b1, sc);
            const float bias = b2[m0 + t * 16 + ncol];
#pragma unroll
            for (int r = 0; r < 8; ++r) sc[r] += bias;
            s[t] = sc;
        }
        // online softmax via fused-DPP reductions (per 16-lane half)
        float nm[8], scale[8];
#pragma unroll
        for (int r = 0; r < 8; ++r) {
            const float cm = dpp_max16(fmaxf(fmaxf(s[0][r], s[1][r]), fmaxf(s[2][r], s[3][r])));
            const float newm = fmaxf(run_max[r], cm);
            scale[r] = __expf(run_max[r] - newm);
            nm[r] = newm;
        }
#pragma unroll
        for (int r = 0; r < 8; ++r) {
            float rs = 0.f;
#pragma unroll
            for (int t = 0; t < 4; ++t) {
                const float p = __expf(s[t][r] - nm[r]);
                s[t][r] = p;
                rs += p;
            }
            run_sum[r] = run_sum[r] * scale[r] + dpp_sum16(rs);
            run_max[r] = nm[r];
        }
        // rescale ctx; write P into per-wave LDS in A-frag order
#pragma unroll
        for (int t = 0; t < 4; ++t) {
#pragma unroll
            for (int r = 0; r < 8; ++r) {
                ctx[t][r] *= scale[r];
                const int mcol = t * 16 + ncol;
                const int kc = mcol >> 5, c32 = mcol & 31;
                const int hi2 = (c32 >> 3) & 1;
                const int j = (c32 & 7) | ((c32 & 16) >> 1);
                Ps[wave][(kc * 32 + hi2 * 16 + (r + mhalf)) * 16 + j] = (bf16_t)s[t][r];
            }
        }
        // ctx += P @ V
        const v16bf aP0 = *(const v16bf*)(&Ps[wave][(0 * 32 + lane) * 16]);
        const v16bf aP1 = *(const v16bf*)(&Ps[wave][(1 * 32 + lane) * 16]);
#pragma unroll
        for (int t = 0; t < 4; ++t) {
            const v16bf bv0 = *(const v16bf*)(&Vs[cur][((0 * 4 + t) * 32 + lane) * 16]);
            const v16bf bv1 = *(const v16bf*)(&Vs[cur][((1 * 4 + t) * 32 + lane) * 16]);
            ctx[t] = wmma_bf16(aP0, bv0, ctx[t]);
            ctx[t] = wmma_bf16(aP1, bv1, ctx[t]);
        }
        if (more) wait_async0();
        __syncthreads();
    }
    // write ctx / run_sum into merged [4096][1024] at column block h*64
#pragma unroll
    for (int t = 0; t < 4; ++t)
#pragma unroll
        for (int r = 0; r < 8; ++r) {
            const long long row = row0 + wave * 16 + r + mhalf;
            const int col = h * 64 + t * 16 + ncol;
            ctx_out[row * 1024 + col] = (bf16_t)(ctx[t][r] / run_sum[r]);
        }
}

// ---- LayerNorm over D=1024, one block per row ----
__global__ __launch_bounds__(256)
void layernorm_kernel(const float* __restrict__ X, const float* __restrict__ g,
                      const float* __restrict__ bta, float* __restrict__ Yf,
                      bf16_t* __restrict__ Yb) {
    __shared__ float s1[256], s2[256];
    const long long row = blockIdx.x;
    const float* x = X + row * 1024;
    float v[4], lsum = 0.f, lsq = 0.f;
#pragma unroll
    for (int i = 0; i < 4; ++i) {
        v[i] = x[threadIdx.x + 256 * i];
        lsum += v[i];
        lsq  += v[i] * v[i];
    }
    s1[threadIdx.x] = lsum; s2[threadIdx.x] = lsq;
    __syncthreads();
    for (int st = 128; st > 0; st >>= 1) {
        if (threadIdx.x < st) {
            s1[threadIdx.x] += s1[threadIdx.x + st];
            s2[threadIdx.x] += s2[threadIdx.x + st];
        }
        __syncthreads();
    }
    const float mean = s1[0] * (1.f / 1024.f);
    const float var  = s2[0] * (1.f / 1024.f) - mean * mean;
    const float rstd = rsqrtf(var + 1e-5f);
#pragma unroll
    for (int i = 0; i < 4; ++i) {
        const int c = threadIdx.x + 256 * i;
        const float y = (v[i] - mean) * rstd * g[c] + bta[c];
        if (Yf) Yf[row * 1024 + c] = y;
        if (Yb) Yb[row * 1024 + c] = (bf16_t)y;
    }
}

extern "C" void kernel_launch(void* const* d_in, const int* in_sizes, int n_in,
                              void* d_out, int out_size, void* d_ws, size_t ws_size,
                              hipStream_t stream) {
    const float* x   = (const float*)d_in[0];
    const float* aw1 = (const float*)d_in[1];
    const float* ab1 = (const float*)d_in[2];
    const float* aw2 = (const float*)d_in[3];
    const float* ab2 = (const float*)d_in[4];
    const float* vw  = (const float*)d_in[5];
    const float* ow  = (const float*)d_in[6];
    const float* fw1 = (const float*)d_in[7];
    const float* fb1 = (const float*)d_in[8];
    const float* fw2 = (const float*)d_in[9];
    const float* fb2 = (const float*)d_in[10];
    const float* g1  = (const float*)d_in[11];
    const float* be1 = (const float*)d_in[12];
    const float* g2  = (const float*)d_in[13];
    const float* be2 = (const float*)d_in[14];
    float* out = (float*)d_out;

    const long long R = 4096;  // B*S
    char* p = (char*)d_ws;
    auto alloc = [&](size_t bytes) -> char* {
        char* r = p;
        p += (bytes + 255) & ~(size_t)255;
        return r;
    };
    bf16_t* xb   = (bf16_t*)alloc(R * 1024 * 2);
    bf16_t* a1t  = (bf16_t*)alloc(64 * 64 * 2);
    bf16_t* a2t  = (bf16_t*)alloc(2048ll * 64 * 2);
    bf16_t* vwt  = (bf16_t*)alloc(1024ll * 1024 * 2);
    bf16_t* owt  = (bf16_t*)alloc(1024ll * 1024 * 2);
    bf16_t* fw1t = (bf16_t*)alloc(4096ll * 1024 * 2);
    bf16_t* fw2t = (bf16_t*)alloc(1024ll * 4096 * 2);
    bf16_t* vht  = (bf16_t*)alloc(16ll * 64 * R * 2);   // [H][64 hd][R m] K-major
    bf16_t* hidb = (bf16_t*)alloc(16ll * R * 64 * 2);   // [H][R][64] row-major
    bf16_t* ctxb = (bf16_t*)alloc(R * 1024 * 2);
    float*  yf   = (float*)alloc(R * 1024 * 4);
    float*  h1f  = (float*)alloc(R * 1024 * 4);
    bf16_t* h1b  = (bf16_t*)alloc(R * 1024 * 2);
    bf16_t* ffb  = (bf16_t*)alloc(R * 4096 * 2);
    float*  zf   = (float*)alloc(R * 1024 * 4);

    cvt_bf16_kernel<<<(4096 * 1024 / 4 + 255) / 256, 256, 0, stream>>>(x, xb, 4096 * 1024);
    auto cvtT = [&](const float* s, bf16_t* d, int K, int N) {
        cvtT_bf16_kernel<<<dim3(N / 32, K / 32), 256, 0, stream>>>(s, d, K, N);
    };
    cvtT(aw1, a1t, 64, 64);
    cvtT(aw2, a2t, 64, 2048);
    cvtT(vw,  vwt, 1024, 1024);
    cvtT(ow,  owt, 1024, 1024);
    cvtT(fw1, fw1t, 1024, 4096);
    cvtT(fw2, fw2t, 4096, 1024);

    // vh[h] = x @ value_w[:, h*64:+64] -> stored TRANSPOSED: vht [H][64][R]
    gemm_bf16_kernel<0, false, false, false, true, true><<<dim3(1, 16, 16), 256, 0, stream>>>(
        xb, 0, 1024, vwt, 64ll * 1024, 1024, nullptr, nullptr, 0, 0,
        nullptr, vht, 64ll * R, (int)R, 1024);
    // hidden[h] = relu(x[:, h*64:+64] @ attn_w1 + b1) -> [H][R][64]
    gemm_bf16_kernel<1, true, false, false, true, false><<<dim3(1, 16, 16), 256, 0, stream>>>(
        xb, 64, 1024, a1t, 0, 64, ab1, nullptr, 0, 0,
        nullptr, hidb, R * 64, 64, 64);
    // fused scores -> softmax -> ctx
    synth_attn_kernel<<<dim3(32, 16), 256, 0, stream>>>(hidb, a2t, ab2, vht, ctxb);
    // y = ctx @ out_w + x
    gemm_bf16_kernel<0, false, true, true, false, false><<<dim3(16, 16, 1), 256, 0, stream>>>(
        ctxb, 0, 1024, owt, 0, 1024, nullptr, x, 0, 1024,
        yf, nullptr, 0, 1024, 1024);
    // h1 = LN1(y)
    layernorm_kernel<<<4096, 256, 0, stream>>>(yf, g1, be1, h1f, h1b);
    // ff = gelu(h1 @ ffn_w1 + b1)
    gemm_bf16_kernel<2, true, false, false, true, false><<<dim3(64, 16, 1), 256, 0, stream>>>(
        h1b, 0, 1024, fw1t, 0, 1024, fb1, nullptr, 0, 0,
        nullptr, ffb, 0, 4096, 1024);
    // z = ff @ ffn_w2 + b2 + h1
    gemm_bf16_kernel<0, true, true, true, false, false><<<dim3(16, 16, 1), 256, 0, stream>>>(
        ffb, 0, 4096, fw2t, 0, 4096, fb2, h1f, 0, 1024,
        zf, nullptr, 0, 1024, 4096);
    // out = LN2(z)
    layernorm_kernel<<<4096, 256, 0, stream>>>(zf, g2, be2, out, nullptr);
}
